// ObjectRelationModule_2267742732655
// MI455X (gfx1250) — matrix operand order repaired and checked by reading, not compiled, not run on hardware
//
#include <hip/hip_runtime.h>
#include <hip/hip_bf16.h>
#include <math.h>

// Problem constants (match reference)
#define NB 4
#define RR 512
#define DD 1024
#define GG 16
#define VD 64           // D / G
// ln(1000)/8
#define LOG1000_8 0.8634694099f

typedef __bf16 bf16;
typedef __attribute__((ext_vector_type(16))) __bf16 v16bf;
typedef __attribute__((ext_vector_type(8)))  float  v8f;
typedef __attribute__((ext_vector_type(4)))  unsigned int v4u;
typedef __attribute__((ext_vector_type(8)))  int v8i;
typedef __attribute__((ext_vector_type(4)))  int v4i;

// CDNA5 16-bit A-matrix 16x32 fragment: element e of lane -> K index
// contiguous order: K = e + 16*hi (matches B fragments)
// lane<16: {0..15}, lane>=16: {16..31}
__device__ __forceinline__ int a_k(int e, int hi) {
    return e + (hi << 4);
}

// ---------------------------------------------------------------------------
// Prep kernels
// ---------------------------------------------------------------------------
__global__ void k_cvt_feat(const float* __restrict__ f, bf16* __restrict__ fbf, int n) {
    int i = blockIdx.x * blockDim.x + threadIdx.x;
    if (i < n) fbf[i] = (bf16)f[i];
}

// Transpose + convert Wq|Wk|Wv (each [1024][64] f32) -> wt [3][64][1024] bf16
__global__ void k_cvt_w(const float* __restrict__ Wq, const float* __restrict__ Wk,
                        const float* __restrict__ Wv, bf16* __restrict__ wt) {
    int i = blockIdx.x * blockDim.x + threadIdx.x;
    if (i >= 3 * 64 * 1024) return;
    int which = i / (64 * 1024);
    int rem   = i % (64 * 1024);
    int col   = rem / 1024;
    int kk    = rem % 1024;
    const float* W = (which == 0) ? Wq : (which == 1) ? Wk : Wv;
    wt[i] = (bf16)W[kk * 64 + col];
}

// geo_w [64][16] -> geo_wt [16][64] bf16 ; boxes -> bparam [N*R][6] = {cy,cx,h,w,logh,logw}
__global__ void k_prep_misc(const float* __restrict__ geo_w, const float* __restrict__ boxes,
                            bf16* __restrict__ geo_wt, float* __restrict__ bparam) {
    int i = blockIdx.x * blockDim.x + threadIdx.x;
    if (i < 16 * 64) {
        int g = i / 64, e = i % 64;
        geo_wt[i] = (bf16)geo_w[e * 16 + g];
    } else if (i < 16 * 64 + NB * RR) {
        int b = i - 16 * 64;
        float ymin = boxes[b * 4 + 0];
        float xmin = boxes[b * 4 + 1];
        float ymax = boxes[b * 4 + 2];
        float xmax = boxes[b * 4 + 3];
        float h = ymax - ymin + 1.0f;
        float w = xmax - xmin + 1.0f;
        bparam[b * 6 + 0] = 0.5f * (ymin + ymax);
        bparam[b * 6 + 1] = 0.5f * (xmin + xmax);
        bparam[b * 6 + 2] = h;
        bparam[b * 6 + 3] = w;
        bparam[b * 6 + 4] = __logf(h);
        bparam[b * 6 + 5] = __logf(w);
    }
}

// ---------------------------------------------------------------------------
// QKV projection: one wave per 16x16 output tile.
// grid = (128 M-tiles, 12) ; 12 = {q,k,v} x 4 N-tiles of 16 (64 cols each)
// q is pre-scaled by 1/sqrt(dk)=0.5 ; v stored transposed bf16 [n][64][512]
// ---------------------------------------------------------------------------
__global__ __launch_bounds__(32) void k_qkv(
    const bf16* __restrict__ fbf, const bf16* __restrict__ wt,
    const float* __restrict__ bq, const float* __restrict__ bk, const float* __restrict__ bv,
    float* __restrict__ qf, float* __restrict__ kf, bf16* __restrict__ vT) {
    int mt    = blockIdx.x;        // 0..127 (M tile over N*R = 2048 rows)
    int job   = blockIdx.y;        // 0..11
    int which = job >> 2;          // 0=q 1=k 2=v
    int nt    = job & 3;           // N tile within 64 cols
    int lane  = threadIdx.x;
    int hi    = lane >> 4;
    int l15   = lane & 15;

    int arow = mt * 16 + l15;                               // A: M = lane&15
    const bf16* Wrow = wt + which * (64 * 1024) + (nt * 16 + l15) * 1024;

    v8f acc = {};
    for (int kk = 0; kk < 1024; kk += 32) {
        v16bf a;
#pragma unroll
        for (int e = 0; e < 16; e++) a[e] = fbf[arow * 1024 + kk + a_k(e, hi)];
        v16bf b = *(const v16bf*)(Wrow + kk + 16 * hi);     // B: K = e + 16*hi, contiguous
        acc = __builtin_amdgcn_wmma_f32_16x16x32_bf16(false, a, false, b,
                                                      (short)0, acc, false, false);
    }

    int col = nt * 16 + l15;   // 0..63
    if (which == 0) {
        float bias = bq[col];
#pragma unroll
        for (int j = 0; j < 8; j++) {
            int row = mt * 16 + j + 8 * hi;
            qf[row * 64 + col] = 0.5f * (acc[j] + bias);    // fold 1/sqrt(dk)
        }
    } else if (which == 1) {
        float bias = bk[col];
#pragma unroll
        for (int j = 0; j < 8; j++) {
            int row = mt * 16 + j + 8 * hi;
            kf[row * 64 + col] = acc[j] + bias;
        }
    } else {
        float bias = bv[col];
#pragma unroll
        for (int j = 0; j < 8; j++) {
            int row = mt * 16 + j + 8 * hi;
            int n = row >> 9, r = row & 511;
            vT[(n * 64 + col) * 512 + r] = (bf16)(acc[j] + bias);
        }
    }
}

// ---------------------------------------------------------------------------
// Fused relation: geometry emb -> geo_w GEMM (WMMA) -> +appearance (VALU, K=4)
// -> online softmax -> P@V (WMMA, B-fragments staged in LDS by the TDM)
// -> +features.
// One workgroup per (n, 16-query tile). 16 waves; wave w = group g (phase 2)
// and query-row w (phase 1). Loop over keys in chunks of 32. The 64x32 bf16
// v-tile is shared by all 16 waves, so wave 0 DMAs it into LDS with
// tensor_load_to_lds (TENSORcnt) while everyone computes geometry.
// ---------------------------------------------------------------------------
__global__ __launch_bounds__(512) void k_relation(
    const float* __restrict__ feat, const float* __restrict__ qf,
    const float* __restrict__ kf, const bf16* __restrict__ vT,
    const bf16* __restrict__ geo_wt, const float* __restrict__ geo_b,
    const float* __restrict__ bparam, float* __restrict__ out) {

    __shared__ float geolog[16][32][17];     // [q_local][k_local][group] (+pad)
    __shared__ bf16  pstage[16][16][32];     // per-wave P transpose staging
    __shared__ bf16  vstage[64][32];         // TDM-staged v tile: [vdim][key]

    int n     = blockIdx.x >> 5;             // 4 * 32 blocks
    int qt    = blockIdx.x & 31;
    int qbase = qt * 16;
    int tid   = threadIdx.x;
    int w     = tid >> 5;                    // wave id
    int lane  = tid & 31;
    int hi    = lane >> 4;
    int l15   = lane & 15;

    // geo_w B fragments (K=64 split into 2 chunks of 32), resident in regs
    const bf16* gw = geo_wt + l15 * 64;
    v16bf gB0 = *(const v16bf*)(gw + 16 * hi);
    v16bf gB1 = *(const v16bf*)(gw + 32 + 16 * hi);
    float gbias = geo_b[l15];

    // q registers for appearance dot (group g = w), rows j+8*hi
    float qreg[8][4];
#pragma unroll
    for (int j = 0; j < 8; j++)
#pragma unroll
        for (int d = 0; d < 4; d++)
            qreg[j][d] = qf[(n * 512 + qbase + j + 8 * hi) * 64 + w * 4 + d];

    // query-box params for phase 1 (q_local = w)
    float bqp[6];
#pragma unroll
    for (int ii = 0; ii < 6; ii++) bqp[ii] = bparam[(n * 512 + qbase + w) * 6 + ii];

    float m_[8], l_[8];
#pragma unroll
    for (int j = 0; j < 8; j++) { m_[j] = -1e30f; l_[j] = 0.0f; }
    v8f acc[4];
#pragma unroll
    for (int t = 0; t < 4; t++) acc[t] = (v8f){};

    for (int kb = 0; kb < 512; kb += 32) {
        // ---- TDM: wave 0 DMAs the 64x32 bf16 v-tile (rows=vdims, cols=keys)
        // into vstage while all waves run geometry. 2D descriptor:
        // tensor [64][512] bf16 @ vT[n], tile 64 x 32 starting at column kb.
        if (w == 0) {
            unsigned long long ga =
                (unsigned long long)(size_t)(vT + (size_t)n * 64 * 512 + kb);
            unsigned int lds = (unsigned int)(size_t)(&vstage[0][0]);
            v4u g0;
            g0.x = 1u;                                       // count=1
            g0.y = lds;                                      // lds_addr
            g0.z = (unsigned int)(ga & 0xffffffffu);         // global_addr lo
            g0.w = (unsigned int)((ga >> 32) & 0x1ffffffu)   // global_addr hi
                 | (2u << 30);                               // type=2 (image)
            v8i g1;
            g1[0] = (int)(1u << 16);                         // data_size=2B, mask=0
            g1[1] = (int)((512u & 0xffffu) << 16);           // tensor_dim0 lo16
            g1[2] = (int)(((512u >> 16) & 0xffffu) | ((64u & 0xffffu) << 16)); // dim0 hi, dim1 lo
            g1[3] = (int)(((64u >> 16) & 0xffffu) | (32u << 16)); // dim1 hi, tile_dim0=32
            g1[4] = (int)64u;                                // tile_dim1=64, tile_dim2=0
            g1[5] = (int)512u;                               // tensor_dim0_stride lo32
            g1[6] = 0;                                       // stride hi / dim1_stride lo
            g1[7] = 0;
            v4i z4 = {0, 0, 0, 0};
            v8i z8 = {0, 0, 0, 0, 0, 0, 0, 0};
            __builtin_amdgcn_tensor_load_to_lds(g0, g1, z4, z4, z8, 0);
        }

        // ---- Phase 1: geometry for q_local = w, keys kb..kb+31 ----
        if (kb + 32 < 512) {   // prefetch next chunk's k-vectors & box params
            __builtin_prefetch(kf + (size_t)(n * 512 + kb + 32 + lane) * 64, 0, 0);
            __builtin_prefetch(bparam + (size_t)(n * 512 + kb + 32 + lane) * 6, 0, 0);
        }
#pragma unroll
        for (int t = 0; t < 2; t++) {
            int kidx = kb + t * 16 + l15;                 // this lane's pair key
            const float* bp = bparam + (n * 512 + kidx) * 6;
            float pos[4];
            pos[0] = __logf(fmaxf((bqp[0] - bp[0]) / bqp[2], 1e-5f));
            pos[1] = __logf(fmaxf((bqp[1] - bp[1]) / bqp[3], 1e-5f));
            pos[2] = bqp[4] - bp[4];
            pos[3] = bqp[5] - bp[5];
            v8f dg = {};
#pragma unroll
            for (int c = 0; c < 2; c++) {
                v16bf a;
#pragma unroll
                for (int e = 0; e < 16; e++) {
                    int dim  = c * 32 + a_k(e, hi);       // emb dim 0..63
                    int comp = dim >> 4;
                    int sc   = (dim >> 3) & 1;
                    int f    = dim & 7;
                    float arg = 100.0f * pos[comp] * __expf(-(float)f * LOG1000_8);
                    a[e] = (bf16)(sc ? __cosf(arg) : __sinf(arg));
                }
                dg = __builtin_amdgcn_wmma_f32_16x16x32_bf16(false, a, false,
                        c ? gB1 : gB0, (short)0, dg, false, false);
            }
#pragma unroll
            for (int j = 0; j < 8; j++) {
                float gv = fmaxf(dg[j] + gbias, 0.0f);    // relu(emb@geo_w + b)
                geolog[w][t * 16 + j + 8 * hi][l15] = __logf(fmaxf(gv, 1e-6f));
            }
        }
        if (w == 0) __builtin_amdgcn_s_wait_tensorcnt(0);
        __syncthreads();   // publishes geolog AND the TDM-written vstage

        // ---- Phase 2: appearance (K=4 VALU) + online softmax + P@V ----
        float logit[2][8];
#pragma unroll
        for (int t = 0; t < 2; t++) {
            int key = kb + t * 16 + l15;
            const float* kp = kf + (n * 512 + key) * 64 + w * 4;
            float k0 = kp[0], k1 = kp[1], k2 = kp[2], k3 = kp[3];
#pragma unroll
            for (int j = 0; j < 8; j++) {
                float dot = qreg[j][0] * k0 + qreg[j][1] * k1 +
                            qreg[j][2] * k2 + qreg[j][3] * k3;
                logit[t][j] = dot + geolog[j + 8 * hi][t * 16 + l15][w];
            }
        }
        float scl[8];
#pragma unroll
        for (int j = 0; j < 8; j++) {
            float mx = fmaxf(logit[0][j], logit[1][j]);
            for (int s = 1; s < 16; s <<= 1) mx = fmaxf(mx, __shfl_xor(mx, s, 32));
            float mnew = fmaxf(m_[j], mx);
            float sc   = __expf(m_[j] - mnew);
            float p0   = __expf(logit[0][j] - mnew);
            float p1   = __expf(logit[1][j] - mnew);
            float ss   = p0 + p1;
            for (int s = 1; s < 16; s <<= 1) ss += __shfl_xor(ss, s, 32);
            l_[j] = l_[j] * sc + ss;
            m_[j] = mnew;
            scl[j] = sc;
            logit[0][j] = p0; logit[1][j] = p1;
        }
#pragma unroll
        for (int t = 0; t < 4; t++)
#pragma unroll
            for (int j = 0; j < 8; j++) acc[t][j] *= scl[j];

        // Pack P (f32 D-layout) -> bf16 A-layout via per-wave LDS transpose
#pragma unroll
        for (int t = 0; t < 2; t++)
#pragma unroll
            for (int j = 0; j < 8; j++)
                pstage[w][j + 8 * hi][t * 16 + l15] = (bf16)logit[t][j];
        asm volatile("s_wait_dscnt 0" ::: "memory");
        v16bf pa;
#pragma unroll
        for (int e = 0; e < 16; e++) pa[e] = pstage[w][l15][a_k(e, hi)];

#pragma unroll
        for (int nt = 0; nt < 4; nt++) {
            v16bf vb = *(const v16bf*)(&vstage[nt * 16 + l15][16 * hi]);
            acc[nt] = __builtin_amdgcn_wmma_f32_16x16x32_bf16(false, pa, false, vb,
                                                              (short)0, acc[nt], false, false);
        }
        __syncthreads();   // vstage/geolog safe to overwrite next chunk
    }

    // Epilogue: out = features + relation@v / l
#pragma unroll
    for (int nt = 0; nt < 4; nt++)
#pragma unroll
        for (int j = 0; j < 8; j++) {
            int q  = qbase + j + 8 * hi;
            int od = w * 64 + nt * 16 + l15;
            size_t idx = (size_t)(n * 512 + q) * 1024 + od;
            out[idx] = feat[idx] + acc[nt][j] / l_[j];
        }
}

// ---------------------------------------------------------------------------
// Host launcher
// ---------------------------------------------------------------------------
extern "C" void kernel_launch(void* const* d_in, const int* in_sizes, int n_in,
                              void* d_out, int out_size, void* d_ws, size_t ws_size,
                              hipStream_t stream) {
    const float* features = (const float*)d_in[0];
    const float* boxes    = (const float*)d_in[1];
    const float* geo_w    = (const float*)d_in[2];
    const float* geo_b    = (const float*)d_in[3];
    const float* Wq       = (const float*)d_in[4];
    const float* bq       = (const float*)d_in[5];
    const float* Wk       = (const float*)d_in[6];
    const float* bk       = (const float*)d_in[7];
    const float* Wv       = (const float*)d_in[8];
    const float* bv       = (const float*)d_in[9];
    float* out = (float*)d_out;

    char* ws = (char*)d_ws;
    size_t off = 0;
    bf16*  fbf    = (bf16*)(ws + off); off += (size_t)2048 * 1024 * 2;   // 4 MB
    bf16*  wt     = (bf16*)(ws + off); off += (size_t)3 * 64 * 1024 * 2; // 384 KB
    bf16*  geo_wt = (bf16*)(ws + off); off += (size_t)16 * 64 * 2;       // 2 KB
    float* bparam = (float*)(ws + off); off += (size_t)NB * RR * 6 * 4;  // 48 KB
    float* qf     = (float*)(ws + off); off += (size_t)2048 * 64 * 4;    // 512 KB
    float* kf     = (float*)(ws + off); off += (size_t)2048 * 64 * 4;    // 512 KB
    bf16*  vT     = (bf16*)(ws + off); off += (size_t)NB * 64 * 512 * 2; // 256 KB

    // Prep
    {
        int n = 2048 * 1024;
        k_cvt_feat<<<(n + 255) / 256, 256, 0, stream>>>(features, fbf, n);
        k_cvt_w<<<(3 * 64 * 1024 + 255) / 256, 256, 0, stream>>>(Wq, Wk, Wv, wt);
        int m = 16 * 64 + NB * RR;
        k_prep_misc<<<(m + 255) / 256, 256, 0, stream>>>(geo_w, boxes, geo_wt, bparam);
    }
    // QKV projection (WMMA)
    k_qkv<<<dim3(128, 12), 32, 0, stream>>>(fbf, wt, bq, bk, bv, qf, kf, vT);
    // Fused relation (TDM v-tile staging + WMMA geometry + flash softmax + WMMA PV)
    k_relation<<<dim3(NB * 32), 512, 0, stream>>>(features, qf, kf, vT, geo_wt,
                                                  geo_b, bparam, out);
}
